// SparseAttention_5712306504197
// MI455X (gfx1250) — matrix operand order repaired and checked by reading, not compiled, not run on hardware
//
#include <hip/hip_runtime.h>
#include <math.h>

// ---------- types ----------
typedef __attribute__((ext_vector_type(16))) __bf16 v16bf;
typedef __attribute__((ext_vector_type(8)))  __bf16 v8bf;
typedef __attribute__((ext_vector_type(8)))  float  v8f;

union FragBF { v16bf v; v8bf h[2]; };

static constexpr int B_  = 2;
static constexpr int S_  = 2048;
static constexpr int H_  = 1024;
static constexpr int NH_ = 16;
static constexpr int HD_ = 64;
static constexpr int M_  = B_ * S_;   // 4096 tokens total

__device__ __forceinline__ __bf16 f2bf(float f) {
  unsigned u = __builtin_bit_cast(unsigned, f);
  unsigned r = u + 0x7FFFu + ((u >> 16) & 1u);      // round-to-nearest-even
  unsigned short hs = (unsigned short)(r >> 16);
  return __builtin_bit_cast(__bf16, hs);
}
__device__ __forceinline__ float bf2f(__bf16 b) {
  unsigned short hs = __builtin_bit_cast(unsigned short, b);
  unsigned u = ((unsigned)hs) << 16;
  return __builtin_bit_cast(float, u);
}

// ---------- fp32 -> bf16 conversion ----------
__global__ void cvt_f32_bf16(const float* __restrict__ in,
                             __bf16* __restrict__ out, int n) {
  int i = blockIdx.x * blockDim.x + threadIdx.x;
  if (i < n) out[i] = f2bf(in[i]);
}

// ---------- GEMM: C[M,N] = A[M,K](bf16) @ W[N,K]^T (bf16) + bias ----------
// W row-major [N,K] is directly the WMMA B operand (col n contiguous over k).
// mode 0: fp32 out row-major [M,N]
// mode 1: bf16 out, per-head layout  [b][h][tok][d]   (Q, K)
// mode 2: bf16 out, per-head transp. [b][h][d][tok]   (V)
__global__ __launch_bounds__(256)
void gemm_bf16_wmma(const __bf16* __restrict__ A, const __bf16* __restrict__ W,
                    const float* __restrict__ bias, void* __restrict__ out,
                    int M, int N, int K, int mode) {
  const int lane = threadIdx.x & 31;
  const int wid  = threadIdx.x >> 5;
  const int rowBase = blockIdx.x * 64  + (wid >> 2) * 32;
  const int colBase = blockIdx.y * 128 + (wid & 3) * 32;
  const int lm  = lane & 15;
  const bool hi = lane >= 16;

  const __bf16* pa0 = A + (size_t)(rowBase + lm) * K + (hi ? 8 : 0);
  const __bf16* pa1 = pa0 + (size_t)16 * K;
  const __bf16* pb0 = W + (size_t)(colBase + lm) * K + (hi ? 8 : 0);
  const __bf16* pb1 = pb0 + (size_t)16 * K;

  v8f c[2][2] = {};
  for (int kk = 0; kk < K; kk += 32) {
    FragBF a[2], b[2];
    a[0].h[0] = *(const v8bf*)(pa0 + kk);
    a[0].h[1] = *(const v8bf*)(pa0 + kk + 16);
    a[1].h[0] = *(const v8bf*)(pa1 + kk);
    a[1].h[1] = *(const v8bf*)(pa1 + kk + 16);
    b[0].h[0] = *(const v8bf*)(pb0 + kk);
    b[0].h[1] = *(const v8bf*)(pb0 + kk + 16);
    b[1].h[0] = *(const v8bf*)(pb1 + kk);
    b[1].h[1] = *(const v8bf*)(pb1 + kk + 16);
    // prefetch next K-step operand lines into cache (global_prefetch_b8)
    if (kk + 32 < K) {
      __builtin_prefetch(pa0 + kk + 32, 0, 3);
      __builtin_prefetch(pa1 + kk + 32, 0, 3);
      __builtin_prefetch(pb0 + kk + 32, 0, 3);
      __builtin_prefetch(pb1 + kk + 32, 0, 3);
    }
#pragma unroll
    for (int ri = 0; ri < 2; ++ri)
#pragma unroll
      for (int ci = 0; ci < 2; ++ci)
        c[ri][ci] = __builtin_amdgcn_wmma_f32_16x16x32_bf16(
            false, a[ri].v, false, b[ci].v, (short)0, c[ri][ci], false, false);
  }
#pragma unroll
  for (int ri = 0; ri < 2; ++ri)
#pragma unroll
    for (int ci = 0; ci < 2; ++ci)
#pragma unroll
      for (int r = 0; r < 8; ++r) {
        int row = rowBase + ri * 16 + (hi ? 8 + r : r);
        int col = colBase + ci * 16 + lm;
        float val = c[ri][ci][r] + bias[col];
        if (mode == 0) {
          ((float*)out)[(size_t)row * N + col] = val;
        } else {
          int bb = row >> 11, tok = row & (S_ - 1);
          int hh = col >> 6,  d   = col & (HD_ - 1);
          size_t idx = (mode == 1)
            ? ((size_t)((bb * NH_ + hh) * S_ + tok)) * HD_ + d
            : ((size_t)((bb * NH_ + hh) * HD_ + d)) * S_ + tok;
          ((__bf16*)out)[idx] = f2bf(val);
        }
      }
}

// ---------- V suffix sums: SufB[b][h][bi][d] = sum_{j >= (bi+1)*32} V[b,h,j,d] ----------
__global__ void suffix_v(const __bf16* __restrict__ Vt, float* __restrict__ SufB) {
  int t = blockIdx.x * blockDim.x + threadIdx.x;   // (b,h,d): 2048 tasks
  if (t >= B_ * NH_ * HD_) return;
  int d = t & 63, bh = t >> 6;
  const __bf16* v = Vt + (size_t)(bh * HD_ + d) * S_;
  float* s = SufB + (size_t)bh * 64 * 64 + d;
  float acc = 0.f;
  for (int bi = 63; bi >= 0; --bi) {
    s[(size_t)bi * 64] = acc;
    int j0 = bi * 32;
#pragma unroll
    for (int j = 0; j < 32; ++j) acc += bf2f(v[j0 + j]);
  }
}

// ---------- block-causal attention with zero-mask fixup ----------
// one wave per (b, h, 16-query tile); 4 waves per block
__global__ __launch_bounds__(128)
void attn_kernel(const __bf16* __restrict__ Qb, const __bf16* __restrict__ Kb,
                 const __bf16* __restrict__ Vt, const float* __restrict__ SufB,
                 __bf16* __restrict__ Ctx) {
  __shared__ __bf16 pShared[4 * 16 * 32];
  const int lane   = threadIdx.x & 31;
  const int wlocal = threadIdx.x >> 5;
  const int w   = blockIdx.x * 4 + wlocal;
  const int qt  = w & 127;          // query tile within (b,h)
  const int bh  = w >> 7;           // 0..31
  const int i0  = qt * 16;
  const int bi  = i0 >> 5;          // 32-wide causal block index
  const int jmax = (bi + 1) * 32;   // allowed keys: j < jmax
  const int lm  = lane & 15;
  const bool hiL = lane >= 16;

  const __bf16* Q  = Qb + (size_t)bh * S_ * HD_;
  const __bf16* Kh = Kb + (size_t)bh * S_ * HD_;
  const __bf16* Vh = Vt + (size_t)bh * HD_ * S_;
  __bf16* lds = pShared + wlocal * 512;

  FragBF aq[2];                                    // Q: 16 rows x 64 d
#pragma unroll
  for (int dc = 0; dc < 2; ++dc) {
    const __bf16* p = Q + (size_t)(i0 + lm) * HD_ + dc * 32 + (hiL ? 8 : 0);
    aq[dc].h[0] = *(const v8bf*)(p);
    aq[dc].h[1] = *(const v8bf*)(p + 16);
  }

  float mrow[8], lrow[8];
#pragma unroll
  for (int r = 0; r < 8; ++r) { mrow[r] = -INFINITY; lrow[r] = 0.f; }
  v8f acc[4] = {};
  const float scale = 0.125f;                      // 1/sqrt(64)

  for (int j0 = 0; j0 < jmax; j0 += 32) {
    // prefetch next chunk's K rows and V strips
    if (j0 + 32 < jmax) {
      __builtin_prefetch(Kh + (size_t)(j0 + 32 + lm) * HD_, 0, 3);
      __builtin_prefetch(Kh + (size_t)(j0 + 48 + lm) * HD_, 0, 3);
      __builtin_prefetch(Vh + (size_t)lm * S_ + j0 + 32, 0, 3);
    }
    v8f s01[2];
#pragma unroll
    for (int jc = 0; jc < 2; ++jc) {               // scores 16x16 per jc
      v8f s = {};
#pragma unroll
      for (int dc = 0; dc < 2; ++dc) {
        FragBF bk;                                 // K^T B frag from row-major K
        const __bf16* p = Kh + (size_t)(j0 + jc * 16 + lm) * HD_ + dc * 32 + (hiL ? 8 : 0);
        bk.h[0] = *(const v8bf*)(p);
        bk.h[1] = *(const v8bf*)(p + 16);
        s = __builtin_amdgcn_wmma_f32_16x16x32_bf16(
            false, aq[dc].v, false, bk.v, (short)0, s, false, false);
      }
      s01[jc] = s;
    }
    // online softmax: per-row reductions across the 16-lane half
    float fac[8];
#pragma unroll
    for (int r = 0; r < 8; ++r) {
      float x = fmaxf(s01[0][r], s01[1][r]) * scale;
      x = fmaxf(x, __shfl_xor(x, 1));
      x = fmaxf(x, __shfl_xor(x, 2));
      x = fmaxf(x, __shfl_xor(x, 4));
      x = fmaxf(x, __shfl_xor(x, 8));
      float mnew = fmaxf(mrow[r], x);
      float f  = __expf(mrow[r] - mnew);
      float p0 = __expf(s01[0][r] * scale - mnew);
      float p1 = __expf(s01[1][r] * scale - mnew);
      float t = p0 + p1;
      t += __shfl_xor(t, 1);
      t += __shfl_xor(t, 2);
      t += __shfl_xor(t, 4);
      t += __shfl_xor(t, 8);
      lrow[r] = lrow[r] * f + t;
      mrow[r] = mnew;
      fac[r]  = f;
      s01[0][r] = p0; s01[1][r] = p1;
    }
#pragma unroll
    for (int fi = 0; fi < 4; ++fi)
#pragma unroll
      for (int r = 0; r < 8; ++r) acc[fi][r] *= fac[r];

    // transpose P (C-layout -> A-layout) through LDS as bf16
#pragma unroll
    for (int r = 0; r < 8; ++r) {
      int row = hiL ? 8 + r : r;
      lds[row * 32 + lm]      = f2bf(s01[0][r]);
      lds[row * 32 + 16 + lm] = f2bf(s01[1][r]);
    }
    asm volatile("s_wait_dscnt 0" ::: "memory");
    FragBF pa;
    {
      const __bf16* p = lds + lm * 32 + (hiL ? 8 : 0);
      pa.h[0] = *(const v8bf*)(p);
      pa.h[1] = *(const v8bf*)(p + 16);
    }
    // ctx += P @ V  (V stored transposed [64, S] => natural B frags)
#pragma unroll
    for (int fi = 0; fi < 4; ++fi) {
      FragBF bv;
      const __bf16* p = Vh + (size_t)(fi * 16 + lm) * S_ + j0 + (hiL ? 8 : 0);
      bv.h[0] = *(const v8bf*)(p);
      bv.h[1] = *(const v8bf*)(p + 16);
      acc[fi] = __builtin_amdgcn_wmma_f32_16x16x32_bf16(
          false, pa.v, false, bv.v, (short)0, acc[fi], false, false);
    }
  }

  // masked region: scores were ZERO (not -inf) -> exp(0-m) fixup with V suffix sum
  const int Nmask = S_ - jmax;
  if (Nmask > 0) {
    float e0[8], fac[8];
#pragma unroll
    for (int r = 0; r < 8; ++r) {
      float mnew = fmaxf(mrow[r], 0.f);
      fac[r] = __expf(mrow[r] - mnew);
      e0[r]  = __expf(-mnew);
      lrow[r] = lrow[r] * fac[r] + (float)Nmask * e0[r];
    }
    const float* suf = SufB + (size_t)bh * 4096 + (size_t)bi * 64;
#pragma unroll
    for (int fi = 0; fi < 4; ++fi) {
      float sv = suf[fi * 16 + lm];
#pragma unroll
      for (int r = 0; r < 8; ++r) acc[fi][r] = acc[fi][r] * fac[r] + e0[r] * sv;
    }
  }

  const int b = bh >> 4, h = bh & 15;
#pragma unroll
  for (int fi = 0; fi < 4; ++fi)
#pragma unroll
    for (int r = 0; r < 8; ++r) {
      int row = b * S_ + i0 + (hiL ? 8 + r : r);
      int col = h * HD_ + fi * 16 + lm;
      Ctx[(size_t)row * H_ + col] = f2bf(acc[fi][r] / lrow[r]);
    }
}

// ---------- launcher ----------
extern "C" void kernel_launch(void* const* d_in, const int* in_sizes, int n_in,
                              void* d_out, int out_size, void* d_ws, size_t ws_size,
                              hipStream_t stream) {
  const float* hs = (const float*)d_in[0];
  const float* Wq = (const float*)d_in[1];  const float* bq = (const float*)d_in[2];
  const float* Wk = (const float*)d_in[3];  const float* bk = (const float*)d_in[4];
  const float* Wv = (const float*)d_in[5];  const float* bv = (const float*)d_in[6];
  const float* Wo = (const float*)d_in[7];  const float* bo = (const float*)d_in[8];

  char* ws = (char*)d_ws;
  const size_t MB = 1024ull * 1024ull;
  __bf16* hsb  = (__bf16*)(ws);                 //  8 MB  [4096,1024]
  __bf16* Wqb  = (__bf16*)(ws +  8 * MB);       //  2 MB
  __bf16* Wkb  = (__bf16*)(ws + 10 * MB);       //  2 MB
  __bf16* Wvb  = (__bf16*)(ws + 12 * MB);       //  2 MB
  __bf16* Wob  = (__bf16*)(ws + 14 * MB);       //  2 MB
  __bf16* Qb   = (__bf16*)(ws + 16 * MB);       //  8 MB  [b][h][tok][d]
  __bf16* Kb   = (__bf16*)(ws + 24 * MB);       //  8 MB  [b][h][tok][d]
  __bf16* Vtb  = (__bf16*)(ws + 32 * MB);       //  8 MB  [b][h][d][tok]
  __bf16* Ctxb = (__bf16*)(ws + 40 * MB);       //  8 MB  [4096,1024]
  float*  SufB = (float*) (ws + 48 * MB);       // .5 MB  [b][h][64][64]

  // 1) convert to bf16
  cvt_f32_bf16<<<(M_ * H_ + 255) / 256, 256, 0, stream>>>(hs, hsb, M_ * H_);
  cvt_f32_bf16<<<(H_ * H_ + 255) / 256, 256, 0, stream>>>(Wq, Wqb, H_ * H_);
  cvt_f32_bf16<<<(H_ * H_ + 255) / 256, 256, 0, stream>>>(Wk, Wkb, H_ * H_);
  cvt_f32_bf16<<<(H_ * H_ + 255) / 256, 256, 0, stream>>>(Wv, Wvb, H_ * H_);
  cvt_f32_bf16<<<(H_ * H_ + 255) / 256, 256, 0, stream>>>(Wo, Wob, H_ * H_);

  // 2) QKV projections (WMMA): block tile 64x128, grid = (M/64, N/128)
  dim3 gg(M_ / 64, H_ / 128);
  gemm_bf16_wmma<<<gg, 256, 0, stream>>>(hsb, Wqb, bq, (void*)Qb,  M_, H_, H_, 1);
  gemm_bf16_wmma<<<gg, 256, 0, stream>>>(hsb, Wkb, bk, (void*)Kb,  M_, H_, H_, 1);
  gemm_bf16_wmma<<<gg, 256, 0, stream>>>(hsb, Wvb, bv, (void*)Vtb, M_, H_, H_, 2);

  // 3) V suffix sums for the zero-mask fixup
  suffix_v<<<(B_ * NH_ * HD_ + 255) / 256, 256, 0, stream>>>(Vtb, SufB);

  // 4) block-causal attention: 4096 wave-tiles, 4 waves/block
  attn_kernel<<<(B_ * NH_ * (S_ / 16)) / 4, 128, 0, stream>>>(Qb, Kb, Vtb, SufB, Ctxb);

  // 5) output projection -> fp32 d_out
  gemm_bf16_wmma<<<gg, 256, 0, stream>>>(Ctxb, Wob, bo, d_out, M_, H_, H_, 0);
}
